// PointNetConv_40192303956568
// MI455X (gfx1250) — matrix operand order repaired and checked by reading, not compiled, not run on hardware
//
#include <hip/hip_runtime.h>
#include <hip/hip_bf16.h>

typedef __bf16 bf16_t;
typedef __attribute__((ext_vector_type(16))) __bf16 v16bf;
typedef __attribute__((ext_vector_type(8)))  __bf16 v8bf;
typedef __attribute__((ext_vector_type(8)))  float  v8f;

#define BN_EPS 1e-5f

// ---------------------------------------------------------------- utilities
__global__ void zero_stats_kernel(float* p, int n) {
    int i = blockIdx.x * blockDim.x + threadIdx.x;
    if (i < n) p[i] = 0.0f;
}

// a = g * rsqrt(var + eps) ; c = beta - a * mean   (conv bias cancels in BN)
__global__ void finalize_kernel(const float* __restrict__ sums,
                                const float* __restrict__ g,
                                const float* __restrict__ be,
                                float* __restrict__ act, int C, float invM) {
    int o = threadIdx.x;
    if (o < C) {
        float mean = sums[o * 2] * invM;
        float var  = sums[o * 2 + 1] * invM - mean * mean;
        float a    = g[o] * rsqrtf(var + BN_EPS);
        act[o * 2]     = a;
        act[o * 2 + 1] = be[o] - a * mean;
    }
}

// out[node,o] = relu(a*(a>=0 ? zmax : zmin) + c)
__global__ void output_kernel(const float* __restrict__ nodeMax,
                              const float* __restrict__ nodeMin,
                              const float* __restrict__ act,
                              float* __restrict__ out, int n) {
    int i = blockIdx.x * blockDim.x + threadIdx.x;
    if (i < n) {
        int o = i & 127;
        float a = act[o * 2];
        float c = act[o * 2 + 1];
        float v = (a >= 0.0f) ? nodeMax[i] : nodeMin[i];
        out[i] = fmaxf(a * v + c, 0.0f);
    }
}

// ---------------------------------------------------------------- GEMM + BN stats
// One layer: Z[M,COUT] = relu_affine(X)[M,CIN] * W^T, accumulating per-channel
// sum / sumsq of Z.  Block = 256 threads = 8 waves; tile = 128 rows x COUT.
// IN_BF16: input is bf16 with per-channel affine+relu (prev layer's BN) on load.
// NODE_RED: instead of storing Z, reduce max/min over groups of 64 rows (nodes).
template <int CIN, int KPAD, int COUT, bool IN_BF16, bool NODE_RED>
__launch_bounds__(256)
__global__ void mlp_layer_kernel(const void* __restrict__ xin,
                                 const float* __restrict__ actIn,  // [CIN][2]
                                 const float* __restrict__ wIn,    // [COUT][CIN] f32
                                 bf16_t* __restrict__ zout,        // [M][COUT]
                                 float* __restrict__ sums,         // [COUT][2]
                                 float* __restrict__ nodeMax,
                                 float* __restrict__ nodeMin,
                                 int M) {
    constexpr int NT = COUT / 16;   // N tiles
    constexpr int KS = KPAD / 32;   // K steps
    constexpr int MT = 128;         // rows per block tile

    __shared__ alignas(16) bf16_t ldsW[COUT * KPAD];
    __shared__ alignas(16) bf16_t ldsA[MT * KPAD];
    __shared__ float ldsStats[COUT * 2];
    __shared__ float ldsActIn[IN_BF16 ? CIN * 2 : 1];
    __shared__ float wredMax[NODE_RED ? 8 * COUT : 1];
    __shared__ float wredMin[NODE_RED ? 8 * COUT : 1];

    const int tid  = threadIdx.x;
    const int lane = tid & 31;
    const int wv   = tid >> 5;    // wave id 0..7
    const int lr   = lane & 15;   // row-in-stripe (A) / col-in-tile (B,C)
    const int hi   = lane >> 4;   // half-wave select

    for (int i = tid; i < COUT * 2; i += 256) ldsStats[i] = 0.0f;
    // zero whole A tile once: K-padding columns stay zero across all tiles
    for (int i = tid; i < MT * KPAD; i += 256) ldsA[i] = (bf16_t)0.0f;
    // stage weights -> LDS bf16, zero-padded in K
    for (int i = tid; i < COUT * KPAD; i += 256) {
        int o = i / KPAD, k = i - o * KPAD;
        ldsW[i] = (bf16_t)((k < CIN) ? wIn[o * CIN + k] : 0.0f);
    }
    if constexpr (IN_BF16) {
        for (int i = tid; i < CIN * 2; i += 256) ldsActIn[i] = actIn[i];
    }
    __syncthreads();

    // B fragments live in registers for the whole kernel.
    // Contiguous-k layout used here (16-bit B 32x16): lane<16 -> col=lane, K=0..15; lane>=16 -> K=16..31.
    v16bf bfrag[NT][KS];
#pragma unroll
    for (int t = 0; t < NT; ++t)
#pragma unroll
        for (int s = 0; s < KS; ++s) {
            const int o  = t * 16 + lr;
            const int kb = s * 32 + hi * 16;
            const v8bf lo = *(const v8bf*)&ldsW[o * KPAD + kb];
            const v8bf hh = *(const v8bf*)&ldsW[o * KPAD + kb + 8];
#pragma unroll
            for (int e = 0; e < 8; ++e) {
                bfrag[t][s][e]     = lo[e];
                bfrag[t][s][e + 8] = hh[e];
            }
        }
    __syncthreads();

    float ssum[NT], ssq[NT];
#pragma unroll
    for (int t = 0; t < NT; ++t) { ssum[t] = 0.0f; ssq[t] = 0.0f; }

    const int numTiles = M / MT;
    for (int tile = blockIdx.x; tile < numTiles; tile += gridDim.x) {
        const int rowBase = tile * MT;

        // ---- stage X tile -> ldsA (bf16) ----
        if constexpr (IN_BF16) {
            // bf16 [M][CIN], CIN==KPAD: vector copy + fused BN-affine + relu
            const bf16_t* x = (const bf16_t*)xin;
            for (int i = tid; i < MT * CIN / 8; i += 256) {
                const int base = i * 8;
                const v8bf src = *(const v8bf*)&x[(size_t)rowBase * CIN + base];
                v8bf dst;
#pragma unroll
                for (int e = 0; e < 8; ++e) {
                    const int k = (base + e) & (CIN - 1);
                    float v = ldsActIn[k * 2] * (float)src[e] + ldsActIn[k * 2 + 1];
                    dst[e] = (bf16_t)fmaxf(v, 0.0f);
                }
                *(v8bf*)&ldsA[base] = dst;
            }
        } else {
            // f32 [M][CIN], CIN=67: coalesced flat stream (non-temporal: one-shot
            // 280MB read; keep L2 free for the resident bf16 intermediates)
            const float* x = (const float*)xin;
            const size_t gbase = (size_t)rowBase * CIN;
            for (int i = tid; i < MT * CIN; i += 256) {
                float v = __builtin_nontemporal_load(&x[gbase + i]);
                int r = i / CIN;
                int k = i - r * CIN;
                ldsA[r * KPAD + k] = (bf16_t)v;
            }
        }
        __syncthreads();

        // ---- WMMA: wave wv computes rows [wv*16, wv*16+16) x all COUT ----
        v8f acc[NT];
#pragma unroll
        for (int t = 0; t < NT; ++t)
#pragma unroll
            for (int e = 0; e < 8; ++e) acc[t][e] = 0.0f;

        // A layout (16-bit A 16x32): lane<16 -> row=lane, K=0..15;
        // lane>=16 -> row=lane-16, K=16..31 (matches B).
        v16bf afrag[KS];
        const int arow = wv * 16 + lr;
#pragma unroll
        for (int s = 0; s < KS; ++s) {
            const int kb = s * 32 + hi * 16;
            const v8bf lo = *(const v8bf*)&ldsA[arow * KPAD + kb];
            const v8bf hh = *(const v8bf*)&ldsA[arow * KPAD + kb + 8];
#pragma unroll
            for (int e = 0; e < 8; ++e) {
                afrag[s][e]     = lo[e];
                afrag[s][e + 8] = hh[e];
            }
        }
#pragma unroll
        for (int t = 0; t < NT; ++t)
#pragma unroll
            for (int s = 0; s < KS; ++s)
                acc[t] = __builtin_amdgcn_wmma_f32_16x16x32_bf16(
                    false, afrag[s], false, bfrag[t][s], (short)0, acc[t],
                    false, false);

        // ---- per-channel BN statistics (register accumulation) ----
#pragma unroll
        for (int t = 0; t < NT; ++t)
#pragma unroll
            for (int e = 0; e < 8; ++e) {
                float v = acc[t][e];
                ssum[t] += v;
                ssq[t]  += v * v;
            }

        if constexpr (!NODE_RED) {
            // C layout: lane holds col=t*16+lr; vgpr e -> row (hi?8:0)+e
            const int rb = rowBase + wv * 16 + hi * 8;
#pragma unroll
            for (int t = 0; t < NT; ++t) {
                const int col = t * 16 + lr;
#pragma unroll
                for (int e = 0; e < 8; ++e)
                    zout[(size_t)(rb + e) * COUT + col] = (bf16_t)acc[t][e];
            }
        } else {
            // reduce max/min over the 64 neighbor rows of each node (2 nodes/tile)
#pragma unroll
            for (int t = 0; t < NT; ++t) {
                float mx = acc[t][0], mn = acc[t][0];
#pragma unroll
                for (int e = 1; e < 8; ++e) {
                    mx = fmaxf(mx, acc[t][e]);
                    mn = fminf(mn, acc[t][e]);
                }
                mx = fmaxf(mx, __shfl_xor(mx, 16, 32));
                mn = fminf(mn, __shfl_xor(mn, 16, 32));
                if (hi == 0) {
                    wredMax[wv * COUT + t * 16 + lr] = mx;
                    wredMin[wv * COUT + t * 16 + lr] = mn;
                }
            }
            __syncthreads();
            {
                const int col  = tid & (COUT - 1);
                const int half = tid / COUT;  // 0 or 1 (COUT==128)
                const int wb   = half * 4;
                float mx = wredMax[wb * COUT + col];
                float mn = wredMin[wb * COUT + col];
#pragma unroll
                for (int w2 = 1; w2 < 4; ++w2) {
                    mx = fmaxf(mx, wredMax[(wb + w2) * COUT + col]);
                    mn = fminf(mn, wredMin[(wb + w2) * COUT + col]);
                }
                const int node = tile * 2 + half;
                nodeMax[(size_t)node * COUT + col] = mx;
                nodeMin[(size_t)node * COUT + col] = mn;
            }
        }
        __syncthreads();
    }

    // ---- flush statistics: registers -> LDS atomics -> 2 global atomics/ch ----
#pragma unroll
    for (int t = 0; t < NT; ++t) {
        const int col = t * 16 + lr;
        atomicAdd(&ldsStats[col * 2],     ssum[t]);
        atomicAdd(&ldsStats[col * 2 + 1], ssq[t]);
    }
    __syncthreads();
    for (int i = tid; i < COUT * 2; i += 256) atomicAdd(&sums[i], ldsStats[i]);
}

// ---------------------------------------------------------------- launch
extern "C" void kernel_launch(void* const* d_in, const int* in_sizes, int n_in,
                              void* d_out, int out_size, void* d_ws, size_t ws_size,
                              hipStream_t stream) {
    const float* agg = (const float*)d_in[0];
    const float* w0  = (const float*)d_in[1];
    const float* g0  = (const float*)d_in[3];
    const float* be0 = (const float*)d_in[4];
    const float* w1  = (const float*)d_in[5];
    const float* g1  = (const float*)d_in[7];
    const float* be1 = (const float*)d_in[8];
    const float* w2  = (const float*)d_in[9];
    const float* g2  = (const float*)d_in[11];
    const float* be2 = (const float*)d_in[12];

    const int nodes = in_sizes[0] / (64 * 67);  // 16384
    const int M     = nodes * 64;               // 1,048,576 positions
    const float invM = 1.0f / (float)M;

    // workspace layout
    char* ws = (char*)d_ws;
    size_t off = 0;
    auto take = [&](size_t bytes) {
        size_t o = off;
        off = (off + bytes + 255) & ~(size_t)255;
        return o;
    };
    bf16_t* Z0      = (bf16_t*)(ws + take((size_t)M * 64 * sizeof(bf16_t)));
    bf16_t* Z1      = (bf16_t*)(ws + take((size_t)M * 64 * sizeof(bf16_t)));
    float*  nodeMax = (float*)(ws + take((size_t)nodes * 128 * sizeof(float)));
    float*  nodeMin = (float*)(ws + take((size_t)nodes * 128 * sizeof(float)));
    float*  sumsAll = (float*)(ws + take(512 * sizeof(float)));
    float*  actAll  = (float*)(ws + take(512 * sizeof(float)));
    float* sums0 = sumsAll, * sums1 = sumsAll + 128, * sums2 = sumsAll + 256;
    float* act0  = actAll,  * act1  = actAll + 128,  * act2  = actAll + 256;

    const int numTiles = M / 128;
    int grid = numTiles < 1024 ? numTiles : 1024;

    zero_stats_kernel<<<2, 256, 0, stream>>>(sumsAll, 512);

    // layer 0: [M,67] f32 -> Z0 [M,64] bf16 (K padded 67->96, 3 k-steps)
    mlp_layer_kernel<67, 96, 64, false, false><<<grid, 256, 0, stream>>>(
        agg, nullptr, w0, Z0, sums0, nullptr, nullptr, M);
    finalize_kernel<<<1, 128, 0, stream>>>(sums0, g0, be0, act0, 64, invM);

    // layer 1: Z0 (+BN0 affine+relu on load) -> Z1 [M,64] bf16
    mlp_layer_kernel<64, 64, 64, true, false><<<grid, 256, 0, stream>>>(
        Z0, act0, w1, Z1, sums1, nullptr, nullptr, M);
    finalize_kernel<<<1, 128, 0, stream>>>(sums1, g1, be1, act1, 64, invM);

    // layer 2: Z1 (+BN1 affine+relu) -> per-node max/min over K=64 (no Z2 store)
    mlp_layer_kernel<64, 64, 128, true, true><<<grid, 256, 0, stream>>>(
        Z1, act1, w2, nullptr, sums2, nodeMax, nodeMin, M);
    finalize_kernel<<<1, 128, 0, stream>>>(sums2, g2, be2, act2, 128, invM);

    // epilogue: out[node,o] = relu(a2 * extreme + c2)
    const int n = nodes * 128;
    output_kernel<<<(n + 255) / 256, 256, 0, stream>>>(nodeMax, nodeMin, act2,
                                                       (float*)d_out, n);
}
